// MFAFuser_86509231276614
// MI455X (gfx1250) — compile-verified, matrix operand order checked by reading, not run on hardware
//
#include <hip/hip_runtime.h>
#include <hip/hip_bf16.h>

// ---------------- constants ----------------
#define IMGC 80
#define PTSC 128
#define INC  208      // IMGC + PTSC
#define INCP 224      // padded to multiple of 32
#define IMGP 96       // IMGC padded
#define DMODEL 256
#define LAYERS 6
#define NHEAD 4
#define NPTS 4
#define NMOD 2
#define HH 128
#define WW 128
#define HWPIX 16384
#define HDIM 64
#define FFND 512
#define CCONV 1024    // DMODEL * S
#define SWEEPS 4

typedef __bf16 bf16_t;
typedef __attribute__((ext_vector_type(16))) __bf16 v16bf;
typedef __attribute__((ext_vector_type(8)))  float  v8f;
typedef __attribute__((ext_vector_type(4)))  unsigned uv4;

union Frag { v16bf v; uv4 q[2]; };

__device__ __forceinline__ bf16_t f2bf(float f) {
    union { float f; unsigned u; } a; a.f = f;
    unsigned r = a.u + 0x7FFFu + ((a.u >> 16) & 1u);   // round-to-nearest-even
    union { unsigned short s; __bf16 h; } b; b.s = (unsigned short)(r >> 16);
    return b.h;
}

// ---------------- weight conversion: src (K x N) f32 -> dst (N x Kpad) bf16 ----------------
__global__ void wconv_kernel(const float* __restrict__ src, bf16_t* __restrict__ dst,
                             int K, int N, int Kpad) {
    int gid = blockIdx.x * 256 + threadIdx.x;
    if (gid >= N * Kpad) return;
    int n = gid / Kpad, k = gid % Kpad;
    dst[gid] = f2bf(k < K ? src[(size_t)k * N + n] : 0.0f);
}

// conv_w (O=256, C=1024, 3, 3) -> dst[t][o][c] bf16  (t = ky*3+kx)
__global__ void convw_kernel(const float* __restrict__ w, bf16_t* __restrict__ dst) {
    int gid = blockIdx.x * 256 + threadIdx.x;     // 9*256*1024
    int t = gid / (DMODEL * CCONV);
    int rem = gid % (DMODEL * CCONV);
    int o = rem / CCONV, c = rem % CCONV;
    dst[gid] = f2bf(w[((size_t)o * CCONV + c) * 9 + t]);
}

// combined bias for so(64)+aw(32) per layer
__global__ void sowab_kernel(const float* __restrict__ so_b, const float* __restrict__ aw_b,
                             float* __restrict__ dst) {
    int gid = blockIdx.x * 64 + threadIdx.x;      // LAYERS*96
    if (gid >= LAYERS * 96) return;
    int l = gid / 96, c = gid % 96;
    dst[gid] = (c < 64) ? so_b[l * 64 + c] : aw_b[l * 32 + (c - 64)];
}

// ---------------- per-pixel channel LayerNorm of input features ----------------
// feats points at one sweep: (208, HW) channel-major
__global__ void ln_feat_kernel(const float* __restrict__ feats,
                               const float* __restrict__ gi, const float* __restrict__ bi,
                               const float* __restrict__ gp, const float* __restrict__ bp,
                               bf16_t* __restrict__ fiA, bf16_t* __restrict__ fpA,
                               bf16_t* __restrict__ fc) {
    int p = blockIdx.x * 256 + threadIdx.x;
    float s = 0.f, s2 = 0.f;
    for (int c = 0; c < IMGC; ++c) { float v = feats[(size_t)c * HWPIX + p]; s += v; s2 += v * v; }
    float m = s * (1.f / IMGC);
    float inv = rsqrtf(s2 * (1.f / IMGC) - m * m + 1e-5f);
    for (int c = 0; c < IMGC; ++c) {
        float v = (feats[(size_t)c * HWPIX + p] - m) * inv * gi[c] + bi[c];
        bf16_t h = f2bf(v);
        fiA[(size_t)p * IMGP + c] = h;
        fc [(size_t)p * INCP + c] = h;
    }
    for (int c = IMGC; c < IMGP; ++c) fiA[(size_t)p * IMGP + c] = f2bf(0.f);
    s = 0.f; s2 = 0.f;
    for (int c = 0; c < PTSC; ++c) { float v = feats[(size_t)(IMGC + c) * HWPIX + p]; s += v; s2 += v * v; }
    m = s * (1.f / PTSC);
    inv = rsqrtf(s2 * (1.f / PTSC) - m * m + 1e-5f);
    for (int c = 0; c < PTSC; ++c) {
        float v = (feats[(size_t)(IMGC + c) * HWPIX + p] - m) * inv * gp[c] + bp[c];
        bf16_t h = f2bf(v);
        fpA[(size_t)p * PTSC + c] = h;
        fc [(size_t)p * INCP + IMGC + c] = h;
    }
    for (int c = INC; c < INCP; ++c) fc[(size_t)p * INCP + c] = f2bf(0.f);
}

// ---------------- qp = bf16(q + pos) ----------------
__global__ void addpos_kernel(const float* __restrict__ q,
                              const float* __restrict__ row_e, const float* __restrict__ col_e,
                              bf16_t* __restrict__ qp) {
    int idx = blockIdx.x * 256 + threadIdx.x;     // HW*256
    int p = idx >> 8, d = idx & 255;
    int x = p & (WW - 1), y = p >> 7;
    float pos = (d < 128) ? col_e[x * 128 + d] : row_e[y * 128 + (d - 128)];
    qp[idx] = f2bf(q[idx] + pos);
}

// ---------------- WMMA bf16 GEMM:  C(MxN) = A(MxK) * Bt(NxK)^T ----------------
// Each wave owns a 32x16 C block (two 16x16 tiles stacked over M) sharing the B
// fragment -> 3 b128 loads per WMMA instead of 4. 8 waves/block => 256 M rows.
__global__ void gemm_bf16_kernel(const bf16_t* __restrict__ A, int lda,
                                 const bf16_t* __restrict__ Bt,
                                 float* __restrict__ outF, bf16_t* __restrict__ outH,
                                 int N, int K, const float* __restrict__ bias, int relu) {
    int wave = threadIdx.x >> 5;
    int lane = threadIdx.x & 31;
    int half = lane >> 4, r = lane & 15;
    int m0 = (blockIdx.y * 8 + wave) * 32;
    int n0 = blockIdx.x * 16;
    v8f c0 = {0.f, 0.f, 0.f, 0.f, 0.f, 0.f, 0.f, 0.f};
    v8f c1 = {0.f, 0.f, 0.f, 0.f, 0.f, 0.f, 0.f, 0.f};
    const bf16_t* ap0 = A  + (size_t)(m0 + r)      * lda + half * 8;
    const bf16_t* ap1 = A  + (size_t)(m0 + 16 + r) * lda + half * 8;
    const bf16_t* bp  = Bt + (size_t)(n0 + r)      * K   + half * 8;
#pragma unroll 4
    for (int k0 = 0; k0 < K; k0 += 32) {
        Frag a0, a1, b;
        b.q[0]  = *(const uv4*)(bp  + k0);
        b.q[1]  = *(const uv4*)(bp  + k0 + 16);
        a0.q[0] = *(const uv4*)(ap0 + k0);
        a0.q[1] = *(const uv4*)(ap0 + k0 + 16);
        a1.q[0] = *(const uv4*)(ap1 + k0);
        a1.q[1] = *(const uv4*)(ap1 + k0 + 16);
        c0 = __builtin_amdgcn_wmma_f32_16x16x32_bf16(false, a0.v, false, b.v,
                                                     (short)0, c0, false, false);
        c1 = __builtin_amdgcn_wmma_f32_16x16x32_bf16(false, a1.v, false, b.v,
                                                     (short)0, c1, false, false);
    }
    int col = n0 + r;
    float bv = bias ? bias[col] : 0.f;
#pragma unroll
    for (int i = 0; i < 8; ++i) {
        int row0 = m0 + half * 8 + i;
        int row1 = row0 + 16;
        float v0 = c0[i] + bv;
        float v1 = c1[i] + bv;
        if (relu) { v0 = fmaxf(v0, 0.f); v1 = fmaxf(v1, 0.f); }
        if (outH) {
            outH[(size_t)row0 * N + col] = f2bf(v0);
            outH[(size_t)row1 * N + col] = f2bf(v1);
        } else {
            outF[(size_t)row0 * N + col] = v0;
            outF[(size_t)row1 * N + col] = v1;
        }
    }
}

// ---------------- softmax + bilinear deformable sampling -> attn (bf16) ----------------
// sowa (HW x 96): cols 0..63 offsets ((h*2+m)*4+pt)*2 + {x,y}; cols 64..95 raw attn logits
__global__ void sample_kernel(const float* __restrict__ sowa,
                              const float* __restrict__ vi, const float* __restrict__ vp,
                              bf16_t* __restrict__ attnb) {
    __shared__ float s_raw[32], s_aw[32], s_px[32], s_py[32];
    int p = blockIdx.x;
    int tid = threadIdx.x;
    if (tid < 32) {
        int h = tid >> 3, sub = tid & 7, m = sub >> 2, pt = sub & 3;
        const float* so = sowa + (size_t)p * 96;
        int oi = ((h * NMOD + m) * NPTS + pt) * 2;
        // ref-point algebra: pixel coord = x + offx (align_corners=False collapses)
        s_px[tid] = (float)(p & (WW - 1)) + so[oi];
        s_py[tid] = (float)(p >> 7)       + so[oi + 1];
        s_raw[tid] = so[64 + h * 8 + sub];
    }
    __syncthreads();
    if (tid < 32) {
        int h = tid >> 3;
        float mx = -1e30f;
        for (int j = 0; j < 8; ++j) mx = fmaxf(mx, s_raw[h * 8 + j]);
        float sum = 0.f;
        for (int j = 0; j < 8; ++j) sum += __expf(s_raw[h * 8 + j] - mx);
        s_aw[tid] = __expf(s_raw[tid] - mx) / sum;
    }
    __syncthreads();
    int h = tid >> 6, d = tid & 63;
    float acc = 0.f;
#pragma unroll
    for (int j2 = 0; j2 < 8; ++j2) {
        int j = h * 8 + j2;
        const float* V = (j2 & 4) ? vp : vi;          // m = j2>>2
        float px = s_px[j], py = s_py[j], aw = s_aw[j];
        float x0f = floorf(px), y0f = floorf(py);
        int x0 = (int)x0f, y0 = (int)y0f;
        float wx = px - x0f, wy = py - y0f;
        size_t ch = (size_t)h * HDIM + d;
        float v00 = 0.f, v01 = 0.f, v10 = 0.f, v11 = 0.f;
        if ((unsigned)x0 < (unsigned)WW && (unsigned)y0 < (unsigned)HH)
            v00 = V[(size_t)(y0 * WW + x0) * DMODEL + ch];
        if ((unsigned)(x0 + 1) < (unsigned)WW && (unsigned)y0 < (unsigned)HH)
            v01 = V[(size_t)(y0 * WW + x0 + 1) * DMODEL + ch];
        if ((unsigned)x0 < (unsigned)WW && (unsigned)(y0 + 1) < (unsigned)HH)
            v10 = V[(size_t)((y0 + 1) * WW + x0) * DMODEL + ch];
        if ((unsigned)(x0 + 1) < (unsigned)WW && (unsigned)(y0 + 1) < (unsigned)HH)
            v11 = V[(size_t)((y0 + 1) * WW + x0 + 1) * DMODEL + ch];
        acc += aw * (v00 * (1.f - wx) * (1.f - wy) + v01 * wx * (1.f - wy) +
                     v10 * (1.f - wx) * wy         + v11 * wx * wy);
    }
    attnb[(size_t)p * DMODEL + tid] = f2bf(acc);
}

// ---------------- q = LN(q + res); also emit bf16 copy (+ optional sweep-output slice) ----------------
__global__ void ln_res_kernel(float* __restrict__ q, const float* __restrict__ res,
                              const float* __restrict__ g, const float* __restrict__ b,
                              bf16_t* __restrict__ qb, bf16_t* __restrict__ qs /* stride 1024 */) {
    int wave = threadIdx.x >> 5, lane = threadIdx.x & 31;
    int p = blockIdx.x * 8 + wave;
    int base = p * DMODEL + lane * 8;
    float v[8];
    float s = 0.f;
#pragma unroll
    for (int i = 0; i < 8; ++i) { v[i] = q[base + i] + res[base + i]; s += v[i]; }
    for (int o = 16; o; o >>= 1) s += __shfl_xor(s, o);
    float m = s * (1.f / DMODEL);
    float s2 = 0.f;
#pragma unroll
    for (int i = 0; i < 8; ++i) { float d = v[i] - m; s2 += d * d; }
    for (int o = 16; o; o >>= 1) s2 += __shfl_xor(s2, o);
    float inv = rsqrtf(s2 * (1.f / DMODEL) + 1e-5f);
#pragma unroll
    for (int i = 0; i < 8; ++i) {
        int c = lane * 8 + i;
        float y = (v[i] - m) * inv * g[c] + b[c];
        q[base + i] = y;
        qb[base + i] = f2bf(y);
        if (qs) qs[(size_t)p * CCONV + c] = f2bf(y);
    }
}

// ---------------- 3x3 conv as implicit WMMA GEMM + BN + ReLU ----------------
// Wt: 9 taps of (O=256 x C=1024) bf16; qs: (HW x 1024) bf16; out: (256 x HW) f32
// Each wave owns two O tiles (rows o0 and o0+16) sharing the pixel (B) fragment.
__global__ void conv_kernel(const bf16_t* __restrict__ Wt, const bf16_t* __restrict__ qs,
                            const float* __restrict__ bn_g, const float* __restrict__ bn_b,
                            float* __restrict__ out) {
    int wave = threadIdx.x >> 5, lane = threadIdx.x & 31;
    int half = lane >> 4, r = lane & 15;
    int o0 = wave * 32;                 // 8 waves * 32 = 256 output channels
    int p0 = blockIdx.x * 16;
    int p = p0 + r;
    int x = p & (WW - 1), y = p >> 7;
    v8f c0 = {0.f, 0.f, 0.f, 0.f, 0.f, 0.f, 0.f, 0.f};
    v8f c1 = {0.f, 0.f, 0.f, 0.f, 0.f, 0.f, 0.f, 0.f};
    for (int t = 0; t < 9; ++t) {
        int dy = t / 3 - 1, dx = t % 3 - 1;
        int xx = x + dx, yy = y + dy;
        bool ok = ((unsigned)xx < (unsigned)WW) && ((unsigned)yy < (unsigned)HH);
        const bf16_t* ap0 = Wt + (size_t)t * DMODEL * CCONV + (size_t)(o0 + r)      * CCONV + half * 8;
        const bf16_t* ap1 = Wt + (size_t)t * DMODEL * CCONV + (size_t)(o0 + 16 + r) * CCONV + half * 8;
        const bf16_t* bp  = qs + (size_t)(yy * WW + xx) * CCONV + half * 8;
#pragma unroll 4
        for (int k0 = 0; k0 < CCONV; k0 += 32) {
            Frag a0, a1, b;
            if (ok) {
                b.q[0] = *(const uv4*)(bp + k0);
                b.q[1] = *(const uv4*)(bp + k0 + 16);
            } else {
                uv4 z = {0u, 0u, 0u, 0u};
                b.q[0] = z; b.q[1] = z;
            }
            a0.q[0] = *(const uv4*)(ap0 + k0);
            a0.q[1] = *(const uv4*)(ap0 + k0 + 16);
            a1.q[0] = *(const uv4*)(ap1 + k0);
            a1.q[1] = *(const uv4*)(ap1 + k0 + 16);
            c0 = __builtin_amdgcn_wmma_f32_16x16x32_bf16(false, a0.v, false, b.v,
                                                         (short)0, c0, false, false);
            c1 = __builtin_amdgcn_wmma_f32_16x16x32_bf16(false, a1.v, false, b.v,
                                                         (short)0, c1, false, false);
        }
    }
    const float sc = rsqrtf(1.0f + 1e-5f);
#pragma unroll
    for (int i = 0; i < 8; ++i) {
        int oA = o0 + half * 8 + i;
        int oB = oA + 16;
        float vA = c0[i] * bn_g[oA] * sc + bn_b[oA];
        float vB = c1[i] * bn_g[oB] * sc + bn_b[oB];
        out[(size_t)oA * HWPIX + p0 + r] = fmaxf(vA, 0.f);
        out[(size_t)oB * HWPIX + p0 + r] = fmaxf(vB, 0.f);
    }
}

// ---------------- host orchestration ----------------
extern "C" void kernel_launch(void* const* d_in, const int* in_sizes, int n_in,
                              void* d_out, int out_size, void* d_ws, size_t ws_size,
                              hipStream_t stream) {
    (void)in_sizes; (void)n_in; (void)out_size; (void)ws_size;
    const float* feats     = (const float*)d_in[0];
    const float* nig       = (const float*)d_in[1];
    const float* nib       = (const float*)d_in[2];
    const float* npg       = (const float*)d_in[3];
    const float* npb       = (const float*)d_in[4];
    const float* in_proj_w = (const float*)d_in[5];
    const float* in_proj_b = (const float*)d_in[6];
    const float* row_embed = (const float*)d_in[7];
    const float* col_embed = (const float*)d_in[8];
    const float* so_w      = (const float*)d_in[9];
    const float* so_b      = (const float*)d_in[10];
    const float* aw_w      = (const float*)d_in[11];
    const float* aw_b      = (const float*)d_in[12];
    const float* vimg_w    = (const float*)d_in[13];
    const float* vimg_b    = (const float*)d_in[14];
    const float* vpts_w    = (const float*)d_in[15];
    const float* vpts_b    = (const float*)d_in[16];
    const float* out_w     = (const float*)d_in[17];
    const float* out_b     = (const float*)d_in[18];
    const float* ffn1_w    = (const float*)d_in[19];
    const float* ffn1_b    = (const float*)d_in[20];
    const float* ffn2_w    = (const float*)d_in[21];
    const float* ffn2_b    = (const float*)d_in[22];
    const float* ln1_g     = (const float*)d_in[23];
    const float* ln1_b     = (const float*)d_in[24];
    const float* ln2_g     = (const float*)d_in[25];
    const float* ln2_b     = (const float*)d_in[26];
    const float* conv_w    = (const float*)d_in[27];
    const float* bn_g      = (const float*)d_in[28];
    const float* bn_b      = (const float*)d_in[29];

    char* w = (char*)d_ws;
    auto alloc = [&](size_t bytes) -> char* {
        char* r = w; w += (bytes + 255) & ~(size_t)255; return r;
    };
    bf16_t* inproj_t = (bf16_t*)alloc((size_t)DMODEL * INCP * 2);
    bf16_t* sowa_t   = (bf16_t*)alloc((size_t)LAYERS * 96 * DMODEL * 2);
    bf16_t* vimg_t   = (bf16_t*)alloc((size_t)LAYERS * DMODEL * IMGP * 2);
    bf16_t* vpts_t   = (bf16_t*)alloc((size_t)LAYERS * DMODEL * PTSC * 2);
    bf16_t* out_t    = (bf16_t*)alloc((size_t)LAYERS * DMODEL * DMODEL * 2);
    bf16_t* ffn1_t   = (bf16_t*)alloc((size_t)LAYERS * FFND * DMODEL * 2);
    bf16_t* ffn2_t   = (bf16_t*)alloc((size_t)LAYERS * DMODEL * FFND * 2);
    bf16_t* convw_t  = (bf16_t*)alloc((size_t)9 * DMODEL * CCONV * 2);
    float*  sowab    = (float*) alloc((size_t)LAYERS * 96 * 4);
    bf16_t* fiA      = (bf16_t*)alloc((size_t)HWPIX * IMGP * 2);
    bf16_t* fpA      = (bf16_t*)alloc((size_t)HWPIX * PTSC * 2);
    bf16_t* fc       = (bf16_t*)alloc((size_t)HWPIX * INCP * 2);
    float*  q        = (float*) alloc((size_t)HWPIX * DMODEL * 4);
    bf16_t* qp       = (bf16_t*)alloc((size_t)HWPIX * DMODEL * 2);
    float*  sowa     = (float*) alloc((size_t)HWPIX * 96 * 4);
    float*  vi       = (float*) alloc((size_t)HWPIX * DMODEL * 4);
    float*  vp       = (float*) alloc((size_t)HWPIX * DMODEL * 4);
    bf16_t* attnb    = (bf16_t*)alloc((size_t)HWPIX * DMODEL * 2);
    float*  gout     = (float*) alloc((size_t)HWPIX * DMODEL * 4);
    bf16_t* h1b      = (bf16_t*)alloc((size_t)HWPIX * FFND * 2);
    bf16_t* qsb      = (bf16_t*)alloc((size_t)HWPIX * CCONV * 2);

    auto wc = [&](const float* src, bf16_t* dst, int K, int N, int Kpad) {
        int n = N * Kpad;
        wconv_kernel<<<(n + 255) / 256, 256, 0, stream>>>(src, dst, K, N, Kpad);
    };

    // ---- weight conversion ----
    wc(in_proj_w, inproj_t, INC, DMODEL, INCP);
    for (int l = 0; l < LAYERS; ++l) {
        wc(so_w + (size_t)l * DMODEL * 64, sowa_t + (size_t)l * 96 * DMODEL, DMODEL, 64, DMODEL);
        wc(aw_w + (size_t)l * DMODEL * 32, sowa_t + (size_t)l * 96 * DMODEL + 64 * DMODEL, DMODEL, 32, DMODEL);
        wc(vimg_w + (size_t)l * IMGC * DMODEL, vimg_t + (size_t)l * DMODEL * IMGP, IMGC, DMODEL, IMGP);
        wc(vpts_w + (size_t)l * PTSC * DMODEL, vpts_t + (size_t)l * DMODEL * PTSC, PTSC, DMODEL, PTSC);
        wc(out_w  + (size_t)l * DMODEL * DMODEL, out_t + (size_t)l * DMODEL * DMODEL, DMODEL, DMODEL, DMODEL);
        wc(ffn1_w + (size_t)l * DMODEL * FFND, ffn1_t + (size_t)l * FFND * DMODEL, DMODEL, FFND, DMODEL);
        wc(ffn2_w + (size_t)l * FFND * DMODEL, ffn2_t + (size_t)l * DMODEL * FFND, FFND, DMODEL, FFND);
    }
    convw_kernel<<<(9 * DMODEL * CCONV) / 256, 256, 0, stream>>>(conv_w, convw_t);
    sowab_kernel<<<(LAYERS * 96 + 63) / 64, 64, 0, stream>>>(so_b, aw_b, sowab);

    auto gemm = [&](const bf16_t* A, int lda, const bf16_t* Bt, float* oF, bf16_t* oH,
                    int N, int K, const float* bias, int relu) {
        dim3 g(N / 16, HWPIX / 256);
        gemm_bf16_kernel<<<g, 256, 0, stream>>>(A, lda, Bt, oF, oH, N, K, bias, relu);
    };

    // ---- sweeps ----
    for (int s = 0; s < SWEEPS; ++s) {
        ln_feat_kernel<<<HWPIX / 256, 256, 0, stream>>>(
            feats + (size_t)s * INC * HWPIX, nig, nib, npg, npb, fiA, fpA, fc);
        gemm(fc, INCP, inproj_t, q, nullptr, DMODEL, INCP, in_proj_b, 0);
        for (int l = 0; l < LAYERS; ++l) {
            addpos_kernel<<<(HWPIX * DMODEL) / 256, 256, 0, stream>>>(q, row_embed, col_embed, qp);
            gemm(qp, DMODEL, sowa_t + (size_t)l * 96 * DMODEL, sowa, nullptr, 96, DMODEL,
                 sowab + l * 96, 0);
            gemm(fiA, IMGP, vimg_t + (size_t)l * DMODEL * IMGP, vi, nullptr, DMODEL, IMGP,
                 vimg_b + (size_t)l * DMODEL, 0);
            gemm(fpA, PTSC, vpts_t + (size_t)l * DMODEL * PTSC, vp, nullptr, DMODEL, PTSC,
                 vpts_b + (size_t)l * DMODEL, 0);
            sample_kernel<<<HWPIX, 256, 0, stream>>>(sowa, vi, vp, attnb);
            gemm(attnb, DMODEL, out_t + (size_t)l * DMODEL * DMODEL, gout, nullptr, DMODEL, DMODEL,
                 out_b + (size_t)l * DMODEL, 0);
            ln_res_kernel<<<HWPIX / 8, 256, 0, stream>>>(
                q, gout, ln1_g + (size_t)l * DMODEL, ln1_b + (size_t)l * DMODEL, qp, nullptr);
            gemm(qp, DMODEL, ffn1_t + (size_t)l * FFND * DMODEL, nullptr, h1b, FFND, DMODEL,
                 ffn1_b + (size_t)l * FFND, 1);
            gemm(h1b, FFND, ffn2_t + (size_t)l * DMODEL * FFND, gout, nullptr, DMODEL, FFND,
                 ffn2_b + (size_t)l * DMODEL, 0);
            ln_res_kernel<<<HWPIX / 8, 256, 0, stream>>>(
                q, gout, ln2_g + (size_t)l * DMODEL, ln2_b + (size_t)l * DMODEL, qp,
                (l == LAYERS - 1) ? (qsb + (size_t)s * DMODEL) : nullptr);
        }
    }

    // ---- 3x3 conv + BN + ReLU ----
    conv_kernel<<<HWPIX / 16, 256, 0, stream>>>(convw_t, qsb, bn_g, bn_b, (float*)d_out);
}
